// Rope_MHA_20899310862922
// MI455X (gfx1250) — matrix operand order repaired and checked by reading, not compiled, not run on hardware
//
#include <hip/hip_runtime.h>
#include <hip/hip_bf16.h>

// ---------------------------------------------------------------------------
// Problem constants (fixed by the reference)
// ---------------------------------------------------------------------------
#define BATCH   2
#define SEQ     2048
#define DMODEL  1024
#define NHEADS  16
#define HDIM    64          // DMODEL / NHEADS
#define MROWS   (BATCH*SEQ) // 4096 GEMM M dimension

typedef __attribute__((ext_vector_type(16))) __bf16 v16bf;
typedef __attribute__((ext_vector_type(8)))  float  v8f;

// ---------------------------------------------------------------------------
// Helpers
// ---------------------------------------------------------------------------
static __device__ __forceinline__ unsigned short f2bf(float f) {
    unsigned int u = __float_as_uint(f);
    u += 0x7FFFu + ((u >> 16) & 1u);      // round-to-nearest-even
    return (unsigned short)(u >> 16);
}

// Build a 16-element bf16 fragment from two 16-byte chunks.
static __device__ __forceinline__ v16bf load_frag2(const unsigned short* p0,
                                                   const unsigned short* p1) {
    uint4 lo = *(const uint4*)p0;
    uint4 hi = *(const uint4*)p1;
    v16bf r;
    __builtin_memcpy(&r, &lo, 16);
    __builtin_memcpy(((char*)&r) + 16, &hi, 16);
    return r;
}

// A-operand (16x32 bf16): lanes 0-15 hold K {0..7,16..23}, lanes 16-31 +8.
static __device__ __forceinline__ v16bf load_fragA(const unsigned short* row, int k0, int hi) {
    const unsigned short* p = row + k0 + 8*hi;
    return load_frag2(p, p + 16);
}
// B-operand (32x16 bf16): lanes 0-15 hold K {0..7,16..23} of their column, lanes 16-31 +8.
static __device__ __forceinline__ v16bf load_fragB(const unsigned short* col, int k0, int hi) {
    const unsigned short* p = col + k0 + 8*hi;
    return load_frag2(p, p + 16);
}

// ---------------------------------------------------------------------------
// 1) f32 -> bf16 convert
// ---------------------------------------------------------------------------
__global__ void rope_mha_cvt_bf16(const float* __restrict__ in,
                                  unsigned short* __restrict__ out, int n) {
    int i = blockIdx.x * blockDim.x + threadIdx.x;
    if (i < n) out[i] = f2bf(in[i]);
}

// ---------------------------------------------------------------------------
// 2) bf16 WMMA GEMM:  C[m,n] = sum_k A[m,k] * Bw[n,k]   (M=4096, N=K=1024)
//    LAYOUT==0 : C row-major f32 [M,N]
//    LAYOUT==1 : C f32 in [b, h, s, d]  (m=b*SEQ+s, n=h*HDIM+d)
//    Block: 128 threads (4 waves), block tile 64(M) x 128(N), wave tile 32x64.
// ---------------------------------------------------------------------------
template <int LAYOUT>
__global__ __launch_bounds__(128)
void rope_mha_gemm_bf16(const unsigned short* __restrict__ A,
                        const unsigned short* __restrict__ Bw,
                        float* __restrict__ C) {
    const int K = DMODEL, N = DMODEL;
    const int wave = threadIdx.x >> 5;
    const int lane = threadIdx.x & 31;
    const int l16  = lane & 15;
    const int hi   = lane >> 4;
    const int M0 = blockIdx.y * 64 + (wave >> 1) * 32;
    const int N0 = blockIdx.x * 128 + (wave & 1) * 64;

    v8f acc[2][4];
#pragma unroll
    for (int mi = 0; mi < 2; ++mi)
#pragma unroll
        for (int ni = 0; ni < 4; ++ni)
#pragma unroll
            for (int r = 0; r < 8; ++r) acc[mi][ni][r] = 0.f;

    for (int k0 = 0; k0 < K; k0 += 32) {
        v16bf a[2], b[4];
#pragma unroll
        for (int mi = 0; mi < 2; ++mi)
            a[mi] = load_fragA(A + (size_t)(M0 + mi*16 + l16) * K, k0, hi);
#pragma unroll
        for (int ni = 0; ni < 4; ++ni)
            b[ni] = load_fragB(Bw + (size_t)(N0 + ni*16 + l16) * K, k0, hi);
#pragma unroll
        for (int mi = 0; mi < 2; ++mi)
#pragma unroll
            for (int ni = 0; ni < 4; ++ni)
                acc[mi][ni] = __builtin_amdgcn_wmma_f32_16x16x32_bf16(
                    false, a[mi], false, b[ni], (short)0, acc[mi][ni], false, false);
    }

#pragma unroll
    for (int mi = 0; mi < 2; ++mi)
#pragma unroll
        for (int ni = 0; ni < 4; ++ni) {
            const int col = N0 + ni*16 + l16;
#pragma unroll
            for (int r = 0; r < 8; ++r) {
                const int row = M0 + mi*16 + r + 8*hi;
                if (LAYOUT == 0) {
                    C[(size_t)row * N + col] = acc[mi][ni][r];
                } else {
                    // [b,h,s,d]: straight-line shift/mask address math
                    const int b_ = row >> 11, s = row & (SEQ - 1);
                    const int h  = col >> 6,  d = col & (HDIM - 1);
                    C[(((size_t)b_ * NHEADS + h) * SEQ + s) * HDIM + d] = acc[mi][ni][r];
                }
            }
        }
}

// ---------------------------------------------------------------------------
// 3) RoPE on raw f32 Q,K ([b,h,s,d]) -> bf16 Q,K (same layout).
//    out[j]    = t[2j]*cos - t[2j+1]*sin
//    out[32+j] = t[2j+1]*cos + t[2j]*sin
// ---------------------------------------------------------------------------
__global__ void rope_mha_rope(const float* __restrict__ qraw,
                              const float* __restrict__ kraw,
                              const float* __restrict__ cosb,
                              const float* __restrict__ sinb,
                              unsigned short* __restrict__ qb,
                              unsigned short* __restrict__ kb) {
    int idx = blockIdx.x * blockDim.x + threadIdx.x;   // [0, B*H*S*32)
    if (idx >= BATCH * NHEADS * SEQ * (HDIM / 2)) return;
    const int j   = idx & 31;
    const int row = idx >> 5;                 // (b*H + h)*SEQ + s
    const int s   = row & (SEQ - 1);
    const size_t base = (size_t)row * HDIM;
    const float c  = cosb[s * 32 + j];
    const float sn = sinb[s * 32 + j];

    float qe = qraw[base + 2*j], qo = qraw[base + 2*j + 1];
    qb[base + j]      = f2bf(qe * c - qo * sn);
    qb[base + 32 + j] = f2bf(qo * c + qe * sn);

    float ke = kraw[base + 2*j], ko = kraw[base + 2*j + 1];
    kb[base + j]      = f2bf(ke * c - ko * sn);
    kb[base + 32 + j] = f2bf(ko * c + ke * sn);
}

// ---------------------------------------------------------------------------
// 4) V raw f32 [b,h,s,d] -> bf16 transposed [b,h,d,s]
// ---------------------------------------------------------------------------
__global__ void rope_mha_vtrans(const float* __restrict__ in,
                                unsigned short* __restrict__ out) {
    int idx = blockIdx.x * blockDim.x + threadIdx.x;   // [0, B*H*S*D)
    if (idx >= BATCH * NHEADS * SEQ * HDIM) return;
    const int bh  = idx >> 17;                // SEQ*HDIM = 131072 = 2^17
    const int rem = idx & 131071;
    const int s   = rem >> 6;
    const int d   = rem & 63;
    out[((size_t)bh * HDIM + d) * SEQ + s] = f2bf(in[idx]);
}

// ---------------------------------------------------------------------------
// 5) Causal flash attention. One wave per 16-query tile per (b,h).
//    Writes ctx bf16 directly in [b, s, h*64+d] row-major layout.
// ---------------------------------------------------------------------------
__global__ __launch_bounds__(128)
void rope_mha_flash(const unsigned short* __restrict__ Qb,
                    const unsigned short* __restrict__ Kb,
                    const unsigned short* __restrict__ Vtb,
                    unsigned short* __restrict__ Ctx) {
    __shared__ unsigned short pbuf[4 * 16 * 32];       // 1 KB per wave

    const int wave = threadIdx.x >> 5;
    const int lane = threadIdx.x & 31;
    const int l16  = lane & 15;
    const int hi   = lane >> 4;
    const int g    = blockIdx.x * 4 + wave;            // 0..4095
    const int bh   = g >> 7;                           // 0..31
    const int qt   = g & 127;
    const int q0   = qt * 16;
    const int b_   = bh >> 4;
    const int h    = bh & 15;

    const unsigned short* qbase = Qb  + (size_t)bh * SEQ * HDIM;
    const unsigned short* kbase = Kb  + (size_t)bh * SEQ * HDIM;
    const unsigned short* vbase = Vtb + (size_t)bh * HDIM * SEQ;
    unsigned short* pb = &pbuf[wave * 512];

    // Q fragments (A operand) for d-steps 0 and 32
    v16bf qa[2];
    {
        const unsigned short* rp = qbase + (size_t)(q0 + l16) * HDIM;
        qa[0] = load_fragA(rp, 0, hi);
        qa[1] = load_fragA(rp, 32, hi);
    }

    float mrow[8], lrow[8];
    v8f acc[4];
#pragma unroll
    for (int r = 0; r < 8; ++r) { mrow[r] = -3.0e38f; lrow[r] = 0.f; }
#pragma unroll
    for (int c = 0; c < 4; ++c)
#pragma unroll
        for (int r = 0; r < 8; ++r) acc[c][r] = 0.f;

    const int ntiles = (q0 + 16 + 31) >> 5;            // 32-key tiles
    for (int t = 0; t < ntiles; ++t) {
        const int t0 = t << 5;

        // S = Q K^T  (16 x 32), two 16-col fragments
        v8f sfr[2];
#pragma unroll
        for (int j = 0; j < 2; ++j) {
            v8f sv;
#pragma unroll
            for (int r = 0; r < 8; ++r) sv[r] = 0.f;
            const unsigned short* kp = kbase + (size_t)(t0 + j*16 + l16) * HDIM;
            v16bf k0 = load_fragB(kp, 0, hi);
            v16bf k1 = load_fragB(kp, 32, hi);
            sv = __builtin_amdgcn_wmma_f32_16x16x32_bf16(false, qa[0], false, k0,
                                                         (short)0, sv, false, false);
            sv = __builtin_amdgcn_wmma_f32_16x16x32_bf16(false, qa[1], false, k1,
                                                         (short)0, sv, false, false);
            sfr[j] = sv;
        }

        // scale + causal mask
#pragma unroll
        for (int j = 0; j < 2; ++j) {
            const int col = t0 + j*16 + l16;
#pragma unroll
            for (int r = 0; r < 8; ++r) {
                const int row = q0 + r + 8*hi;
                float v = sfr[j][r] * 0.125f;          // 1/sqrt(64)
                sfr[j][r] = (col > row) ? -3.0e38f : v;
            }
        }

        // online softmax (rows live across 16 lanes of a half-wave)
        float prob0[8], prob1[8], alphas[8];
#pragma unroll
        for (int r = 0; r < 8; ++r) {
            float mt = fmaxf(sfr[0][r], sfr[1][r]);
#pragma unroll
            for (int d = 1; d < 16; d <<= 1) mt = fmaxf(mt, __shfl_xor(mt, d, 32));
            const float mn = fmaxf(mrow[r], mt);
            const float alpha = __expf(mrow[r] - mn);
            const float p0 = __expf(sfr[0][r] - mn);
            const float p1 = __expf(sfr[1][r] - mn);
            float sum = p0 + p1;
#pragma unroll
            for (int d = 1; d < 16; d <<= 1) sum += __shfl_xor(sum, d, 32);
            lrow[r] = lrow[r] * alpha + sum;
            mrow[r] = mn;
            prob0[r] = p0; prob1[r] = p1; alphas[r] = alpha;
        }
#pragma unroll
        for (int c = 0; c < 4; ++c)
#pragma unroll
            for (int r = 0; r < 8; ++r) acc[c][r] *= alphas[r];

        // P (C-layout) -> LDS -> A-operand fragment (per-wave slice, in-wave
        // DS ordering guaranteed; s_wait_dscnt orders store->load)
#pragma unroll
        for (int r = 0; r < 8; ++r) {
            pb[(r + 8*hi) * 32 + l16]      = f2bf(prob0[r]);
            pb[(r + 8*hi) * 32 + 16 + l16] = f2bf(prob1[r]);
        }
        asm volatile("s_wait_dscnt 0x0" ::: "memory");
        v16bf pa = load_fragA(pb + l16 * 32, 0, hi);

        // acc += P * V   (V^T is [d, s] so B columns are contiguous)
#pragma unroll
        for (int c = 0; c < 4; ++c) {
            const unsigned short* vp = vbase + (size_t)(c*16 + l16) * SEQ + t0;
            v16bf vb = load_fragB(vp, 0, hi);
            acc[c] = __builtin_amdgcn_wmma_f32_16x16x32_bf16(false, pa, false, vb,
                                                             (short)0, acc[c], false, false);
        }
    }

    // normalize and write ctx bf16 in [b, s, dmodel]
#pragma unroll
    for (int r = 0; r < 8; ++r) {
        const float inv = 1.f / lrow[r];
        const int row = q0 + r + 8*hi;
        const size_t base = ((size_t)b_ * SEQ + row) * DMODEL + h * HDIM;
#pragma unroll
        for (int c = 0; c < 4; ++c)
            Ctx[base + c*16 + l16] = f2bf(acc[c][r] * inv);
    }
}

// ---------------------------------------------------------------------------
// Launch
// ---------------------------------------------------------------------------
extern "C" void kernel_launch(void* const* d_in, const int* in_sizes, int n_in,
                              void* d_out, int out_size, void* d_ws, size_t ws_size,
                              hipStream_t stream) {
    const float* x    = (const float*)d_in[0];
    const float* wq   = (const float*)d_in[1];
    const float* wk   = (const float*)d_in[2];
    const float* wv   = (const float*)d_in[3];
    const float* wo   = (const float*)d_in[4];
    const float* cosb = (const float*)d_in[5];
    const float* sinb = (const float*)d_in[6];

    char* ws = (char*)d_ws;
    const size_t MB = (size_t)1 << 20;
    unsigned short* xb   = (unsigned short*)(ws +  0 * MB);  // 8 MB
    unsigned short* wqb  = (unsigned short*)(ws +  8 * MB);  // 2 MB
    unsigned short* wkb  = (unsigned short*)(ws + 10 * MB);
    unsigned short* wvb  = (unsigned short*)(ws + 12 * MB);
    unsigned short* wob  = (unsigned short*)(ws + 14 * MB);
    float*          qraw = (float*)(ws + 16 * MB);           // 16 MB
    float*          kraw = (float*)(ws + 32 * MB);
    float*          vraw = (float*)(ws + 48 * MB);
    unsigned short* qb   = (unsigned short*)(ws + 64 * MB);  // 8 MB
    unsigned short* kb   = (unsigned short*)(ws + 72 * MB);
    unsigned short* vtb  = (unsigned short*)(ws + 80 * MB);
    unsigned short* ctxb = (unsigned short*)(ws + 88 * MB);

    const int NX = MROWS * DMODEL;      // 4194304
    const int NW = DMODEL * DMODEL;     // 1048576

    rope_mha_cvt_bf16<<<(NX + 255) / 256, 256, 0, stream>>>(x,  xb,  NX);
    rope_mha_cvt_bf16<<<(NW + 255) / 256, 256, 0, stream>>>(wq, wqb, NW);
    rope_mha_cvt_bf16<<<(NW + 255) / 256, 256, 0, stream>>>(wk, wkb, NW);
    rope_mha_cvt_bf16<<<(NW + 255) / 256, 256, 0, stream>>>(wv, wvb, NW);
    rope_mha_cvt_bf16<<<(NW + 255) / 256, 256, 0, stream>>>(wo, wob, NW);

    dim3 gg(DMODEL / 128, MROWS / 64), gb(128);
    rope_mha_gemm_bf16<1><<<gg, gb, 0, stream>>>(xb, wqb, qraw);
    rope_mha_gemm_bf16<1><<<gg, gb, 0, stream>>>(xb, wkb, kraw);
    rope_mha_gemm_bf16<1><<<gg, gb, 0, stream>>>(xb, wvb, vraw);

    const int NP = BATCH * NHEADS * SEQ * (HDIM / 2);   // 2097152
    rope_mha_rope<<<(NP + 255) / 256, 256, 0, stream>>>(qraw, kraw, cosb, sinb, qb, kb);
    rope_mha_vtrans<<<(NX + 255) / 256, 256, 0, stream>>>(vraw, vtb);

    rope_mha_flash<<<(BATCH * NHEADS * (SEQ / 16)) / 4, 128, 0, stream>>>(qb, kb, vtb, ctxb);

    rope_mha_gemm_bf16<0><<<gg, gb, 0, stream>>>(ctxb, wob, (float*)d_out);
}
